// Qwen2TransformerBlock_89404039233894
// MI455X (gfx1250) — matrix operand order repaired and checked by reading, not compiled, not run on hardware
//
#include <hip/hip_runtime.h>
#include <hip/hip_bf16.h>

typedef __attribute__((ext_vector_type(16))) __bf16 v16bf;
typedef __attribute__((ext_vector_type(8)))  float  v8f;
typedef int v4i __attribute__((vector_size(16)));   // matches builtin's int4 pointee

#define WMMA_BF16(a, b, c) \
  __builtin_amdgcn_wmma_f32_16x16x32_bf16(false, (a), false, (b), (short)0, (c), false, false)

union FragBF { v16bf v; uint4 q[2]; };

// ---- CDNA5 async global->LDS copy (ASYNCcnt), builtin if available else ISA asm
__device__ __forceinline__ void async_ld_b128(void* lds, const void* g) {
#if __has_builtin(__builtin_amdgcn_global_load_async_to_lds_b128)
  __builtin_amdgcn_global_load_async_to_lds_b128(
      (__attribute__((address_space(1))) v4i*)(unsigned long long)g,
      (__attribute__((address_space(3))) v4i*)(unsigned int)(unsigned long long)lds,
      0, 0);
#else
  unsigned int l32 = (unsigned int)(unsigned long long)lds;      // LDS offset = generic addr[31:0]
  unsigned long long g64 = (unsigned long long)g;
  asm volatile("global_load_async_to_lds_b128 %0, %1, off"
               :: "v"(l32), "v"(g64) : "memory");
#endif
}

template <int N>
__device__ __forceinline__ void async_wait() {
#if __has_builtin(__builtin_amdgcn_s_wait_asynccnt)
  __builtin_amdgcn_s_wait_asynccnt(N);
#else
  asm volatile("s_wait_asynccnt %0" :: "i"(N) : "memory");
#endif
}

// ---------------------------------------------------------------- fp32 -> bf16
__global__ void cvt_f32_bf16(const float* __restrict__ src, __bf16* __restrict__ dst, long n) {
  long i = (long)blockIdx.x * blockDim.x + threadIdx.x;
  long stride = (long)gridDim.x * blockDim.x;
  for (; i < n; i += stride) dst[i] = (__bf16)src[i];
}

// ---------------------------------------------------------------- RMSNorm -> bf16
__global__ __launch_bounds__(256) void rmsnorm_bf16(const float* __restrict__ x,
                                                    const float* __restrict__ w,
                                                    __bf16* __restrict__ out, int K) {
  int t = blockIdx.x;
  const float* row = x + (size_t)t * K;
  float ss = 0.f;
  for (int j = threadIdx.x; j < K; j += 256) { float v = row[j]; ss += v * v; }
  #pragma unroll
  for (int m = 16; m >= 1; m >>= 1) ss += __shfl_xor(ss, m, 32);
  __shared__ float red[8];
  if ((threadIdx.x & 31) == 0) red[threadIdx.x >> 5] = ss;
  __syncthreads();
  float tot = 0.f;
  #pragma unroll
  for (int i = 0; i < 8; ++i) tot += red[i];
  float rn = rsqrtf(tot / (float)K + 1e-6f);
  __bf16* orow = out + (size_t)t * K;
  for (int j = threadIdx.x; j < K; j += 256) orow[j] = (__bf16)(row[j] * rn * w[j]);
}

// ---------------------------------------------------------------- NT GEMM: C = A[M,K] * W[N,K]^T
// mode 0: outB = bf16(C);  mode 1: outF = C + res
__global__ __launch_bounds__(256) void gemm_nt_bf16(const __bf16* __restrict__ A,
                                                    const __bf16* __restrict__ W,
                                                    __bf16* __restrict__ outB,
                                                    float* __restrict__ outF,
                                                    const float* __restrict__ res,
                                                    int Mi, int Ni, int Ki, int mode) {
  __shared__ __align__(16) __bf16 AsB[2 * 128 * 32];
  __shared__ __align__(16) __bf16 BsB[2 * 128 * 32];
  int tid = threadIdx.x, wid = tid >> 5, lane = tid & 31, lr = lane & 15, hi = lane >> 4;
  int wm = wid >> 1, wn = wid & 1;
  int m0 = blockIdx.y * 128, n0 = blockIdx.x * 128;

  auto issueTile = [&](int k0, int buf) {
    __bf16* As = AsB + buf * (128 * 32);
    __bf16* Bs = BsB + buf * (128 * 32);
    #pragma unroll
    for (int c0 = 0; c0 < 2; ++c0) {
      int c = tid + c0 * 256;
      int row = c >> 2, col = (c & 3) << 3;
      async_ld_b128(&As[row * 32 + col], &A[(size_t)(m0 + row) * Ki + k0 + col]);
      async_ld_b128(&Bs[row * 32 + col], &W[(size_t)(n0 + row) * Ki + k0 + col]);
    }
  };

  v8f acc[2][4] = {};
  int nk = Ki >> 5;
  issueTile(0, 0);
  for (int kt = 0; kt < nk; ++kt) {
    if (kt + 1 < nk) { issueTile((kt + 1) << 5, (kt + 1) & 1); async_wait<4>(); }
    else             { async_wait<0>(); }
    __syncthreads();

    const __bf16* As = AsB + (kt & 1) * (128 * 32);
    const __bf16* Bs = BsB + (kt & 1) * (128 * 32);
    FragBF a[2], b[4];
    #pragma unroll
    for (int i = 0; i < 2; ++i) {
      int r = wm * 32 + i * 16 + lr;
      a[i].q[0] = *(const uint4*)&As[r * 32 + hi * 8];
      a[i].q[1] = *(const uint4*)&As[r * 32 + 16 + hi * 8];
    }
    #pragma unroll
    for (int j = 0; j < 4; ++j) {
      int r = wn * 64 + j * 16 + lr;
      b[j].q[0] = *(const uint4*)&Bs[r * 32 + hi * 8];
      b[j].q[1] = *(const uint4*)&Bs[r * 32 + 16 + hi * 8];
    }
    #pragma unroll
    for (int i = 0; i < 2; ++i)
      #pragma unroll
      for (int j = 0; j < 4; ++j)
        acc[i][j] = WMMA_BF16(a[i].v, b[j].v, acc[i][j]);
    __syncthreads();
  }
  #pragma unroll
  for (int i = 0; i < 2; ++i)
    #pragma unroll
    for (int j = 0; j < 4; ++j)
      #pragma unroll
      for (int r = 0; r < 8; ++r) {
        int mm = m0 + wm * 32 + i * 16 + r + 8 * hi;
        int nn = n0 + wn * 64 + j * 16 + lr;
        size_t idx = (size_t)mm * Ni + nn;
        if (mode == 0) outB[idx] = (__bf16)acc[i][j][r];
        else           outF[idx] = acc[i][j][r] + res[idx];
      }
}

// ---------------------------------------------------------------- fused gate/up GEMM + SiLU*up
__global__ __launch_bounds__(256) void gateup_kernel(const __bf16* __restrict__ A,
                                                     const __bf16* __restrict__ Wg,
                                                     const __bf16* __restrict__ Wu,
                                                     __bf16* __restrict__ outb,
                                                     int Mi, int Ni, int Ki) {
  __shared__ __align__(16) __bf16 AsB[2 * 128 * 32];
  __shared__ __align__(16) __bf16 GsB[2 * 64 * 32];
  __shared__ __align__(16) __bf16 UsB[2 * 64 * 32];
  int tid = threadIdx.x, wid = tid >> 5, lane = tid & 31, lr = lane & 15, hi = lane >> 4;
  int wm = wid >> 1, wn = wid & 1;
  int m0 = blockIdx.y * 128, n0 = blockIdx.x * 64;

  auto issueTile = [&](int k0, int buf) {
    __bf16* As = AsB + buf * (128 * 32);
    __bf16* Gs = GsB + buf * (64 * 32);
    __bf16* Us = UsB + buf * (64 * 32);
    #pragma unroll
    for (int c0 = 0; c0 < 2; ++c0) {
      int c = tid + c0 * 256;
      int row = c >> 2, col = (c & 3) << 3;
      async_ld_b128(&As[row * 32 + col], &A[(size_t)(m0 + row) * Ki + k0 + col]);
    }
    {
      int row = tid >> 2, col = (tid & 3) << 3;
      async_ld_b128(&Gs[row * 32 + col], &Wg[(size_t)(n0 + row) * Ki + k0 + col]);
      async_ld_b128(&Us[row * 32 + col], &Wu[(size_t)(n0 + row) * Ki + k0 + col]);
    }
  };

  v8f cg[2][2] = {}, cu[2][2] = {};
  int nk = Ki >> 5;
  issueTile(0, 0);
  for (int kt = 0; kt < nk; ++kt) {
    if (kt + 1 < nk) { issueTile((kt + 1) << 5, (kt + 1) & 1); async_wait<4>(); }
    else             { async_wait<0>(); }
    __syncthreads();

    const __bf16* As = AsB + (kt & 1) * (128 * 32);
    const __bf16* Gs = GsB + (kt & 1) * (64 * 32);
    const __bf16* Us = UsB + (kt & 1) * (64 * 32);
    FragBF a[2], g[2], u[2];
    #pragma unroll
    for (int i = 0; i < 2; ++i) {
      int r = wm * 32 + i * 16 + lr;
      a[i].q[0] = *(const uint4*)&As[r * 32 + hi * 8];
      a[i].q[1] = *(const uint4*)&As[r * 32 + 16 + hi * 8];
    }
    #pragma unroll
    for (int j = 0; j < 2; ++j) {
      int r = wn * 32 + j * 16 + lr;
      g[j].q[0] = *(const uint4*)&Gs[r * 32 + hi * 8];
      g[j].q[1] = *(const uint4*)&Gs[r * 32 + 16 + hi * 8];
      u[j].q[0] = *(const uint4*)&Us[r * 32 + hi * 8];
      u[j].q[1] = *(const uint4*)&Us[r * 32 + 16 + hi * 8];
    }
    #pragma unroll
    for (int i = 0; i < 2; ++i)
      #pragma unroll
      for (int j = 0; j < 2; ++j) {
        cg[i][j] = WMMA_BF16(a[i].v, g[j].v, cg[i][j]);
        cu[i][j] = WMMA_BF16(a[i].v, u[j].v, cu[i][j]);
      }
    __syncthreads();
  }
  #pragma unroll
  for (int i = 0; i < 2; ++i)
    #pragma unroll
    for (int j = 0; j < 2; ++j)
      #pragma unroll
      for (int r = 0; r < 8; ++r) {
        int mm = m0 + wm * 32 + i * 16 + r + 8 * hi;
        int nn = n0 + wn * 32 + j * 16 + lr;
        float gv = cg[i][j][r], uv = cu[i][j][r];
        outb[(size_t)mm * Ni + nn] = (__bf16)(gv / (1.f + __expf(-gv)) * uv);
      }
}

// ---------------------------------------------------------------- RoPE + repack q/k -> [b,h,l,d], v -> [b,h,d,l]
__global__ __launch_bounds__(256) void rope_pack(const __bf16* __restrict__ qb,
                                                 const __bf16* __restrict__ kb,
                                                 const __bf16* __restrict__ vb,
                                                 __bf16* __restrict__ qr,
                                                 __bf16* __restrict__ kr,
                                                 __bf16* __restrict__ vt,
                                                 const int* __restrict__ offp) {
  const int L = 2048, NH = 16, D = 128;
  int t = blockIdx.x;
  int b = t >> 11, l = t & 2047;
  int off = offp[0];
  size_t tok = (size_t)t * 2048;
  const float C = 0.14391156544f;  // ln(10000)/64
  #pragma unroll
  for (int pp = 0; pp < 4; ++pp) {
    int p = threadIdx.x + pp * 256;
    int h = p >> 6, i = p & 63;
    float ang = (float)(l + off) * __expf(-(float)i * C);
    float cs = cosf(ang), sn = sinf(ang);
    size_t src = tok + (size_t)h * D;
    size_t dst = (((size_t)(b * NH + h)) * L + l) * D;
    float x1 = (float)qb[src + i], x2 = (float)qb[src + 64 + i];
    qr[dst + i]      = (__bf16)(x1 * cs - x2 * sn);
    qr[dst + 64 + i] = (__bf16)(x2 * cs + x1 * sn);
    x1 = (float)kb[src + i]; x2 = (float)kb[src + 64 + i];
    kr[dst + i]      = (__bf16)(x1 * cs - x2 * sn);
    kr[dst + 64 + i] = (__bf16)(x2 * cs + x1 * sn);
  }
  #pragma unroll
  for (int pp = 0; pp < 8; ++pp) {
    int d2 = threadIdx.x + pp * 256;
    int h = d2 >> 7, d = d2 & 127;
    vt[((size_t)(b * NH + h) * D + d) * L + l] = vb[tok + d2];
  }
}

// ---------------------------------------------------------------- flash attention (causal), WMMA QK^T and P*V
__global__ __launch_bounds__(256) void attn_kernel(const __bf16* __restrict__ qr,
                                                   const __bf16* __restrict__ kr,
                                                   const __bf16* __restrict__ vt,
                                                   __bf16* __restrict__ outb) {
  __shared__ __align__(16) __bf16 KsB[2 * 32 * 128];
  __shared__ __align__(16) __bf16 VsB[2 * 128 * 32];
  __shared__ __align__(16) __bf16 Ps[8 * 16 * 32];
  const int L = 2048, D = 128, H = 2048;
  int bh = blockIdx.y;
  int q0 = blockIdx.x * 128;
  int tid = threadIdx.x, w = tid >> 5, lane = tid & 31, lr = lane & 15, hi = lane >> 4;
  const __bf16* qbase = qr + (size_t)bh * L * D;
  const __bf16* kbase = kr + (size_t)bh * L * D;
  const __bf16* vbase = vt + (size_t)bh * D * L;

  auto issueKV = [&](int kv0, int buf) {
    __bf16* Ks = KsB + buf * (32 * 128);
    __bf16* Vs = VsB + buf * (128 * 32);
    #pragma unroll
    for (int c0 = 0; c0 < 2; ++c0) {
      int c = tid + c0 * 256;
      { int row = c >> 4, col = (c & 15) << 3;
        async_ld_b128(&Ks[row * 128 + col], &kbase[(size_t)(kv0 + row) * D + col]); }
      { int row = c >> 2, col = (c & 3) << 3;
        async_ld_b128(&Vs[row * 32 + col], &vbase[(size_t)row * L + kv0 + col]); }
    }
  };

  FragBF qf[4];
  int qrow = q0 + w * 16 + lr;
  #pragma unroll
  for (int ks = 0; ks < 4; ++ks) {
    qf[ks].q[0] = *(const uint4*)&qbase[(size_t)qrow * D + ks * 32 + hi * 8];
    qf[ks].q[1] = *(const uint4*)&qbase[(size_t)qrow * D + ks * 32 + 16 + hi * 8];
  }
  v8f o[8] = {};
  float m_r[8], l_r[8];
  #pragma unroll
  for (int r = 0; r < 8; ++r) { m_r[r] = -1e30f; l_r[r] = 0.f; }
  int myq = q0 + w * 16 + 8 * hi;
  int nkt = (q0 + 128) >> 5;

  issueKV(0, 0);
  for (int kt = 0; kt < nkt; ++kt) {
    int kv0 = kt * 32;
    if (kt + 1 < nkt) { issueKV((kt + 1) * 32, (kt + 1) & 1); async_wait<4>(); }
    else              { async_wait<0>(); }
    __syncthreads();

    const __bf16* Ks = KsB + (kt & 1) * (32 * 128);
    const __bf16* Vs = VsB + (kt & 1) * (128 * 32);
    v8f s[2] = {};
    #pragma unroll
    for (int j = 0; j < 2; ++j) {
      int r = j * 16 + lr;
      #pragma unroll
      for (int ks = 0; ks < 4; ++ks) {
        FragBF bk;
        bk.q[0] = *(const uint4*)&Ks[r * 128 + ks * 32 + hi * 8];
        bk.q[1] = *(const uint4*)&Ks[r * 128 + ks * 32 + 16 + hi * 8];
        s[j] = WMMA_BF16(qf[ks].v, bk.v, s[j]);
      }
    }
    const float scale = 0.08838834764831845f;  // 1/sqrt(128)
    #pragma unroll
    for (int r = 0; r < 8; ++r) {
      int qg = myq + r;
      float p0 = (kv0 + lr      <= qg) ? s[0][r] * scale : -1e30f;
      float p1 = (kv0 + 16 + lr <= qg) ? s[1][r] * scale : -1e30f;
      float mx = fmaxf(p0, p1);
      #pragma unroll
      for (int msk = 1; msk < 16; msk <<= 1) mx = fmaxf(mx, __shfl_xor(mx, msk, 32));
      float mnew = fmaxf(m_r[r], mx);
      float sc = __expf(m_r[r] - mnew);
      m_r[r] = mnew;
      float e0 = __expf(p0 - mnew), e1 = __expf(p1 - mnew);
      float rs = e0 + e1;
      #pragma unroll
      for (int msk = 1; msk < 16; msk <<= 1) rs += __shfl_xor(rs, msk, 32);
      l_r[r] = l_r[r] * sc + rs;
      #pragma unroll
      for (int jn = 0; jn < 8; ++jn) o[jn][r] *= sc;
      int prow = r + 8 * hi;
      Ps[(w * 16 + prow) * 32 + lr]      = (__bf16)e0;
      Ps[(w * 16 + prow) * 32 + 16 + lr] = (__bf16)e1;
    }
    asm volatile("s_wait_dscnt 0" ::: "memory");  // intra-wave LDS P-transpose visibility
    FragBF pa;
    pa.q[0] = *(const uint4*)&Ps[(w * 16 + lr) * 32 + hi * 8];
    pa.q[1] = *(const uint4*)&Ps[(w * 16 + lr) * 32 + 16 + hi * 8];
    #pragma unroll
    for (int jn = 0; jn < 8; ++jn) {
      FragBF bv;
      int r = jn * 16 + lr;
      bv.q[0] = *(const uint4*)&Vs[r * 32 + hi * 8];
      bv.q[1] = *(const uint4*)&Vs[r * 32 + 16 + hi * 8];
      o[jn] = WMMA_BF16(pa.v, bv.v, o[jn]);
    }
    __syncthreads();
  }
  int b = bh >> 4, h = bh & 15;
  #pragma unroll
  for (int r = 0; r < 8; ++r) {
    int qg = myq + r;
    float inv = 1.f / l_r[r];
    size_t rowbase = ((size_t)(b * L + qg)) * H + (size_t)h * D;
    #pragma unroll
    for (int jn = 0; jn < 8; ++jn)
      outb[rowbase + jn * 16 + lr] = (__bf16)(o[jn][r] * inv);
  }
}

// ---------------------------------------------------------------- host
extern "C" void kernel_launch(void* const* d_in, const int* in_sizes, int n_in,
                              void* d_out, int out_size, void* d_ws, size_t ws_size,
                              hipStream_t stream) {
  const float* x   = (const float*)d_in[0];
  const float* ln1 = (const float*)d_in[1];
  const float* ln2 = (const float*)d_in[2];
  const float* wq  = (const float*)d_in[3];
  const float* wk  = (const float*)d_in[4];
  const float* wv  = (const float*)d_in[5];
  const float* wo  = (const float*)d_in[6];
  const float* wg  = (const float*)d_in[7];
  const float* wu  = (const float*)d_in[8];
  const float* wd  = (const float*)d_in[9];
  const int*   off = (const int*)d_in[11];
  float* out = (float*)d_out;

  const int Mi = 4096, Hh = 2048, Ii = 8192;
  char* p = (char*)d_ws;
  auto carve = [&](size_t bytes) -> char* {
    char* r = p; p += (bytes + 255) & ~(size_t)255; return r;
  };
  __bf16* wqB = (__bf16*)carve((size_t)Hh * Hh * 2);
  __bf16* wkB = (__bf16*)carve((size_t)Hh * Hh * 2);
  __bf16* wvB = (__bf16*)carve((size_t)Hh * Hh * 2);
  __bf16* woB = (__bf16*)carve((size_t)Hh * Hh * 2);
  __bf16* wgB = (__bf16*)carve((size_t)Ii * Hh * 2);
  __bf16* wuB = (__bf16*)carve((size_t)Ii * Hh * 2);
  __bf16* wdB = (__bf16*)carve((size_t)Hh * Ii * 2);
  __bf16* hb  = (__bf16*)carve((size_t)Mi * Hh * 2);   // reused for h2 after attention
  __bf16* qb  = (__bf16*)carve((size_t)Mi * Hh * 2);   // reused as attn output
  __bf16* kb  = (__bf16*)carve((size_t)Mi * Hh * 2);
  __bf16* vb  = (__bf16*)carve((size_t)Mi * Hh * 2);
  __bf16* qR  = (__bf16*)carve((size_t)Mi * Hh * 2);
  __bf16* kR  = (__bf16*)carve((size_t)Mi * Hh * 2);
  __bf16* vT  = (__bf16*)carve((size_t)Mi * Hh * 2);
  float*  x1  = (float*)carve((size_t)Mi * Hh * 4);
  __bf16* gact = (__bf16*)carve((size_t)Mi * Ii * 2);

  cvt_f32_bf16<<<2048, 256, 0, stream>>>(wq, wqB, (long)Hh * Hh);
  cvt_f32_bf16<<<2048, 256, 0, stream>>>(wk, wkB, (long)Hh * Hh);
  cvt_f32_bf16<<<2048, 256, 0, stream>>>(wv, wvB, (long)Hh * Hh);
  cvt_f32_bf16<<<2048, 256, 0, stream>>>(wo, woB, (long)Hh * Hh);
  cvt_f32_bf16<<<4096, 256, 0, stream>>>(wg, wgB, (long)Ii * Hh);
  cvt_f32_bf16<<<4096, 256, 0, stream>>>(wu, wuB, (long)Ii * Hh);
  cvt_f32_bf16<<<4096, 256, 0, stream>>>(wd, wdB, (long)Hh * Ii);

  rmsnorm_bf16<<<4096, 256, 0, stream>>>(x, ln1, hb, Hh);

  dim3 gH(Hh / 128, Mi / 128);  // (16, 32)
  gemm_nt_bf16<<<gH, 256, 0, stream>>>(hb, wqB, qb, nullptr, nullptr, Mi, Hh, Hh, 0);
  gemm_nt_bf16<<<gH, 256, 0, stream>>>(hb, wkB, kb, nullptr, nullptr, Mi, Hh, Hh, 0);
  gemm_nt_bf16<<<gH, 256, 0, stream>>>(hb, wvB, vb, nullptr, nullptr, Mi, Hh, Hh, 0);

  rope_pack<<<4096, 256, 0, stream>>>(qb, kb, vb, qR, kR, vT, off);

  attn_kernel<<<dim3(16, 32), 256, 0, stream>>>(qR, kR, vT, qb);  // qb reused as attn out

  gemm_nt_bf16<<<gH, 256, 0, stream>>>(qb, woB, nullptr, x1, x, Mi, Hh, Hh, 1);

  rmsnorm_bf16<<<4096, 256, 0, stream>>>(x1, ln2, hb, Hh);

  gateup_kernel<<<dim3(Ii / 64, Mi / 128), 256, 0, stream>>>(hb, wgB, wuB, gact, Mi, Ii, Hh);

  gemm_nt_bf16<<<gH, 256, 0, stream>>>(gact, wdB, nullptr, out, x1, Mi, Hh, Ii, 1);

  (void)in_sizes; (void)n_in; (void)out_size; (void)ws_size;
}
